// MoELayer_82497731821709
// MI455X (gfx1250) — matrix operand order repaired and checked by reading, not compiled, not run on hardware
//
#include <hip/hip_runtime.h>
#include <hip/hip_bf16.h>
#include <math.h>

// ---------------- problem constants (from reference) ----------------
#define NTOK   2048
#define DM     1024     // d_model
#define DFF    4096     // d_ff
#define KPATHS 4
#define RANK   16

// ---------------- types ----------------
typedef __attribute__((ext_vector_type(16))) __bf16 v16bf;
typedef __attribute__((ext_vector_type(8)))  float  v8f;

struct alignas(16) US8 { unsigned short s[8]; };   // one b128 of 8 bf16

__device__ __forceinline__ unsigned short f2bf(float f) {
    unsigned u = __float_as_uint(f);
    u += 0x7FFFu + ((u >> 16) & 1u);              // round-to-nearest-even
    return (unsigned short)(u >> 16);
}

// Branchless tanh-GELU (jax.nn.gelu approximate): lowers to v_exp_f32, no exec divergence.
__device__ __forceinline__ float gelu_tanh(float x) {
    float z = 0.7978845608028654f * (x + 0.044715f * x * x * x);
    z = fminf(fmaxf(z, -15.f), 15.f);
    float e = __expf(2.f * z);
    float th = (e - 1.f) / (e + 1.f);
    return 0.5f * x * (1.f + th);
}

// 8 WMMAs for a 32x64 wave tile (2 M-frags x 4 N-frags)
__device__ __forceinline__ void mma8(v8f acc[2][4], v16bf a0, v16bf a1, const v16bf b[4]) {
#pragma unroll
    for (int j = 0; j < 4; ++j) {
        acc[0][j] = __builtin_amdgcn_wmma_f32_16x16x32_bf16(
                        false, a0, false, b[j], (short)0, acc[0][j], false, false);
        acc[1][j] = __builtin_amdgcn_wmma_f32_16x16x32_bf16(
                        false, a1, false, b[j], (short)0, acc[1][j], false, false);
    }
}

// ---------------- software-pipelined 32x64 wave-tile GEMM core ----------------
// Ping-pong double buffering: fragment sets p/q alternate across a 64-wide K
// step, so loads write directly into the registers consumed one full step
// later (no rotation movs). All addresses are base + compile-time immediate
// (LD, tile offsets fit the signed 24-bit IOFFSET), so no per-iteration
// 64-bit address adds.
// ISA fragment layouts (CDNA5 ISA), bf16 16x16x32:
//   A 16x32: lane L holds M=L&15; e0..7 -> K = kk + 8*(L>=16) + e,
//            e8..15 -> K = kk + 16 + 8*(L>=16) + (e-8)   (hbA folded into Abase)
//   B 32x16: lane L holds N=L&15; e0..7 -> K = kk + 8*(L>=16) + e,
//            e8..15 -> K = kk + 16 + 8*(L>=16) + (e-8)   (hbB folded)
template<int LD, int KT>
__device__ __forceinline__ void gemm_core(v8f acc[2][4],
                                          const unsigned short* __restrict__ Ab,
                                          const unsigned short* __restrict__ Bb) {
#define LDA_FRAG(kk, mi) ({                                                  \
        union { v16bf v; US8 h[2]; } u_;                                     \
        u_.h[0] = *(const US8*)(Ab + (size_t)(mi) * 16 * LD + (kk));         \
        u_.h[1] = *(const US8*)(Ab + (size_t)(mi) * 16 * LD + (kk) + 16);    \
        u_.v; })
#define LDB_FRAG(kk, j) ({                                                   \
        union { v16bf v; US8 h[2]; } u_;                                     \
        u_.h[0] = *(const US8*)(Bb + (size_t)(j) * 16 * LD + (kk));          \
        u_.h[1] = *(const US8*)(Bb + (size_t)(j) * 16 * LD + (kk) + 16);     \
        u_.v; })

    v16bf pa0 = LDA_FRAG(0, 0), pa1 = LDA_FRAG(0, 1);
    v16bf pb[4];
#pragma unroll
    for (int j = 0; j < 4; ++j) pb[j] = LDB_FRAG(0, j);
    v16bf qa0 = LDA_FRAG(32, 0), qa1 = LDA_FRAG(32, 1);
    v16bf qb[4];
#pragma unroll
    for (int j = 0; j < 4; ++j) qb[j] = LDB_FRAG(32, j);

#pragma unroll 1
    for (int kk = 64; kk < KT; kk += 64) {
        mma8(acc, pa0, pa1, pb);
        pa0 = LDA_FRAG(kk, 0);
        pa1 = LDA_FRAG(kk, 1);
#pragma unroll
        for (int j = 0; j < 4; ++j) pb[j] = LDB_FRAG(kk, j);
        mma8(acc, qa0, qa1, qb);
        qa0 = LDA_FRAG(kk + 32, 0);
        qa1 = LDA_FRAG(kk + 32, 1);
#pragma unroll
        for (int j = 0; j < 4; ++j) qb[j] = LDB_FRAG(kk + 32, j);
    }
    mma8(acc, pa0, pa1, pb);
    mma8(acc, qa0, qa1, qb);
#undef LDA_FRAG
#undef LDB_FRAG
}

// ---------------- pack tokens f32 -> bf16 ----------------
__global__ void pack_tokens(const float* __restrict__ tok, unsigned short* __restrict__ xbf) {
    unsigned t = blockIdx.x * blockDim.x + threadIdx.x;
    if (t < NTOK * DM) xbf[t] = f2bf(tok[t]);
}

// ---------------- merged weight build (TT contraction over rank) ----------------
// W1t[k][xy(4096)][ab(1024)] = sum_r c1[k,a,x,r] * c2[k,r,b,y]   (column-major GEMM B)
__global__ void build_w1t(const float* __restrict__ c1, const float* __restrict__ c2,
                          unsigned short* __restrict__ w) {
    unsigned tid = blockIdx.x * blockDim.x + threadIdx.x;  // 4*4096*1024
    if (tid >= (unsigned)KPATHS * DFF * DM) return;
    int k  = tid >> 22;
    int rem = tid & 0x3FFFFF;
    int xy = rem >> 10;               // x*64+y
    int ab = rem & 1023;              // a*32+b
    int a = ab >> 5, b = ab & 31, x = xy >> 6, y = xy & 63;
    const float* p1 = c1 + (((size_t)k * 32 + a) * 64 + x) * 16;       // [k,a,x,:]
    const float* p2 = c2 + (size_t)k * (RANK * 32 * 64) + b * 64 + y;  // [k,r,b,y], r-stride 2048
    float s = 0.f;
#pragma unroll
    for (int r = 0; r < RANK; ++r) s += p1[r] * p2[(size_t)r * 2048];
    w[tid] = f2bf(s);
}

// W2t[k][xy2(1024)][ab2(4096)] = sum_r c1[k,a,x,r] * c2[k,r,b,y]
__global__ void build_w2t(const float* __restrict__ c1, const float* __restrict__ c2,
                          unsigned short* __restrict__ w) {
    unsigned tid = blockIdx.x * blockDim.x + threadIdx.x;  // 4*1024*4096
    if (tid >= (unsigned)KPATHS * DM * DFF) return;
    int k  = tid >> 22;
    int rem = tid & 0x3FFFFF;
    int xy = rem >> 12;               // x*32+y  (output d index)
    int ab = rem & 4095;              // a*64+b  (d_ff index)
    int a = ab >> 6, b = ab & 63, x = xy >> 5, y = xy & 31;
    const float* p1 = c1 + (((size_t)k * 64 + a) * 32 + x) * 16;       // [k,a,x,:]
    const float* p2 = c2 + (size_t)k * (RANK * 64 * 32) + b * 32 + y;  // [k,r,b,y], r-stride 2048
    float s = 0.f;
#pragma unroll
    for (int r = 0; r < RANK; ++r) s += p1[r] * p2[(size_t)r * 2048];
    w[tid] = f2bf(s);
}

// ---------------- routing gates: softmax(tokens @ bases^T) ----------------
__global__ __launch_bounds__(128) void gates_kernel(const float* __restrict__ tok,
                                                    const float* __restrict__ bases,
                                                    float* __restrict__ gates) {
    int n = blockIdx.x;
    int t = threadIdx.x, kp = t >> 5, lane = t & 31;
    const float* tr = tok + (size_t)n * DM;
    const float* br = bases + (size_t)kp * DM;
    float p = 0.f;
    for (int d = lane; d < DM; d += 32) p += tr[d] * br[d];
    __shared__ float s[128];
    __shared__ float lg[KPATHS];
    s[t] = p;
    __syncthreads();
    for (int off = 16; off > 0; off >>= 1) {
        if (lane < off) s[t] += s[t + off];
        __syncthreads();
    }
    if (lane == 0) lg[kp] = s[t];
    __syncthreads();
    if (t == 0) {
        float m = lg[0];
        for (int i = 1; i < KPATHS; ++i) m = fmaxf(m, lg[i]);
        float e[KPATHS], sum = 0.f;
        for (int i = 0; i < KPATHS; ++i) { e[i] = __expf(lg[i] - m); sum += e[i]; }
        float inv = 1.f / sum;
        for (int i = 0; i < KPATHS; ++i) gates[n * KPATHS + i] = e[i] * inv;
    }
}

// ---------------- GEMM1 + tanh-GELU: h[k] = gelu(X @ W1[k]) ----------------
__global__ __launch_bounds__(128) void ffn1_gemm_gelu(const unsigned short* __restrict__ xbf,
                                                      const unsigned short* __restrict__ w1t,
                                                      unsigned short* __restrict__ hbuf) {
    const int lane = threadIdx.x & 31, wave = threadIdx.x >> 5;
    const int kp   = blockIdx.z;
    const int row0 = blockIdx.y * 32;
    const int col0 = (blockIdx.x * 4 + wave) * 64;
    const int rl   = lane & 15;
    const int hbA  = (lane & 16) ? 8 : 0;
    const int hbB  = (lane & 16) ? 8 : 0;

    const unsigned short* Ab = xbf + (size_t)(row0 + rl) * DM + hbA;
    const unsigned short* Bb = w1t + ((size_t)kp * DFF + col0 + rl) * DM + hbB;

    v8f acc[2][4];
#pragma unroll
    for (int mi = 0; mi < 2; ++mi)
#pragma unroll
        for (int j = 0; j < 4; ++j) acc[mi][j] = (v8f){};

    gemm_core<DM, DM>(acc, Ab, Bb);

    const int chi = (lane & 16) ? 8 : 0;
    unsigned short* hk = hbuf + (size_t)kp * NTOK * DFF;
#pragma unroll
    for (int mi = 0; mi < 2; ++mi)
#pragma unroll
        for (int j = 0; j < 4; ++j) {
            int d = col0 + j * 16 + rl;
#pragma unroll
            for (int v = 0; v < 8; ++v) {
                int n = row0 + mi * 16 + chi + v;
                hk[(size_t)n * DFF + d] = f2bf(gelu_tanh(acc[mi][j][v]));
            }
        }
}

// ---------------- GEMM2 + holographic combine over K paths ----------------
__global__ __launch_bounds__(128) void ffn2_gemm_combine(const unsigned short* __restrict__ hbuf,
                                                         const unsigned short* __restrict__ w2t,
                                                         const float* __restrict__ gates,
                                                         const float* __restrict__ pw,
                                                         float* __restrict__ out) {
    const int lane = threadIdx.x & 31, wave = threadIdx.x >> 5;
    const int row0 = blockIdx.y * 32;
    const int col0 = (blockIdx.x * 4 + wave) * 64;
    const int rl   = lane & 15;
    const int hbA  = (lane & 16) ? 8 : 0;
    const int hbB  = (lane & 16) ? 8 : 0;
    const int chi  = (lane & 16) ? 8 : 0;

    float ov[2][4][8];
#pragma unroll
    for (int mi = 0; mi < 2; ++mi)
#pragma unroll
        for (int j = 0; j < 4; ++j)
#pragma unroll
            for (int v = 0; v < 8; ++v) ov[mi][j][v] = 0.f;

#pragma unroll 1
    for (int kpath = 0; kpath < KPATHS; ++kpath) {
        const unsigned short* Ab =
            hbuf + (size_t)kpath * NTOK * DFF + (size_t)(row0 + rl) * DFF + hbA;
        const unsigned short* Bb =
            w2t + ((size_t)kpath * DM + col0 + rl) * DFF + hbB;

        v8f acc[2][4];
#pragma unroll
        for (int mi = 0; mi < 2; ++mi)
#pragma unroll
            for (int j = 0; j < 4; ++j) acc[mi][j] = (v8f){};

        gemm_core<DFF, DFF>(acc, Ab, Bb);

#pragma unroll
        for (int j = 0; j < 4; ++j) {
            float mod = 1.0f + pw[kpath * DM + col0 + j * 16 + rl];
#pragma unroll
            for (int mi = 0; mi < 2; ++mi)
#pragma unroll
                for (int v = 0; v < 8; ++v) {
                    int n = row0 + mi * 16 + chi + v;
                    ov[mi][j][v] += acc[mi][j][v] * mod * gates[n * KPATHS + kpath];
                }
        }
    }

#pragma unroll
    for (int mi = 0; mi < 2; ++mi)
#pragma unroll
        for (int j = 0; j < 4; ++j) {
            int d = col0 + j * 16 + rl;
#pragma unroll
            for (int v = 0; v < 8; ++v)
                out[(size_t)(row0 + mi * 16 + chi + v) * DM + d] = ov[mi][j][v];
        }
}

// ---------------- host side ----------------
extern "C" void kernel_launch(void* const* d_in, const int* in_sizes, int n_in,
                              void* d_out, int out_size, void* d_ws, size_t ws_size,
                              hipStream_t stream) {
    const float* tokens = (const float*)d_in[0];
    const float* f1c1   = (const float*)d_in[1];   // [4,32,64,16]
    const float* f1c2   = (const float*)d_in[2];   // [4,16,32,64]
    const float* f2c1   = (const float*)d_in[3];   // [4,64,32,16]
    const float* f2c2   = (const float*)d_in[4];   // [4,16,64,32]
    const float* bases  = (const float*)d_in[5];   // [4,1024]
    const float* pw     = (const float*)d_in[6];   // [4,1024]
    float* out = (float*)d_out;

    // workspace layout (bytes)
    char* ws = (char*)d_ws;
    const size_t SZ_X  = (size_t)NTOK * DM * 2;              //  4 MB  bf16 tokens
    const size_t SZ_W1 = (size_t)KPATHS * DFF * DM * 2;      // 33.5 MB bf16 W1^T
    const size_t SZ_W2 = (size_t)KPATHS * DM * DFF * 2;      // 33.5 MB bf16 W2^T
    const size_t SZ_H  = (size_t)KPATHS * NTOK * DFF * 2;    // 67 MB  bf16 hidden
    unsigned short* Xbf = (unsigned short*)(ws);
    unsigned short* W1t = (unsigned short*)(ws + SZ_X);
    unsigned short* W2t = (unsigned short*)(ws + SZ_X + SZ_W1);
    unsigned short* H   = (unsigned short*)(ws + SZ_X + SZ_W1 + SZ_W2);
    float*          G   = (float*)(ws + SZ_X + SZ_W1 + SZ_W2 + SZ_H);

    // 1) convert tokens to bf16
    pack_tokens<<<(NTOK * DM + 255) / 256, 256, 0, stream>>>(tokens, Xbf);
    // 2) merge TT cores into dense (column-major) bf16 weight matrices
    build_w1t<<<(KPATHS * DFF * DM + 255) / 256, 256, 0, stream>>>(f1c1, f1c2, W1t);
    build_w2t<<<(KPATHS * DM * DFF + 255) / 256, 256, 0, stream>>>(f2c1, f2c2, W2t);
    // 3) routing gates
    gates_kernel<<<NTOK, 128, 0, stream>>>(tokens, bases, G);
    // 4) h[k] = gelu(X @ W1[k])   — WMMA bf16, 4 waves/block, 32x64 tile/wave
    ffn1_gemm_gelu<<<dim3(DFF / 256, NTOK / 32, KPATHS), 128, 0, stream>>>(Xbf, W1t, H);
    // 5) out = sum_k (h[k] @ W2[k]) * (1+pw[k]) * gate[:,k]   — combine fused
    ffn2_gemm_combine<<<dim3(DM / 256, NTOK / 32, 1), 128, 0, stream>>>(H, W2t, G, pw, out);
}
